// SubNonLocalAttention_60945585930431
// MI455X (gfx1250) — matrix operand order repaired and checked by reading, not compiled, not run on hardware
//
#include <hip/hip_runtime.h>

typedef __bf16 bf16;
typedef float v8f   __attribute__((ext_vector_type(8)));
typedef bf16  v16bf __attribute__((ext_vector_type(16)));

#define CC 64
#define HH 256
#define WW 256
#define NB 2
#define KSEL 5

static __device__ __forceinline__ v8f vzero() {
  v8f z;
#pragma unroll
  for (int i = 0; i < 8; ++i) z[i] = 0.f;
  return z;
}

static __device__ __forceinline__ v8f wmma_bf16(v16bf a, v16bf b, v8f c) {
  // D = A(16x32 bf16) * B(32x16 bf16) + C(16x16 f32)
  return __builtin_amdgcn_wmma_f32_16x16x32_bf16(false, a, false, b, (short)0, c,
                                                 false, false);
}

// ---------------------------------------------------------------------------
// Repack conv weights: w[co][ci][3][3] f32 -> wp[tap][co][ci] bf16
// ---------------------------------------------------------------------------
__global__ void k_pack_w(const float* __restrict__ w, bf16* __restrict__ wp) {
  int i = blockIdx.x * 256 + threadIdx.x;
  if (i >= 9 * CC * CC) return;
  int t = i / (CC * CC), r = i % (CC * CC), co = r / CC, ci = r % CC;
  wp[i] = (bf16)w[(co * CC + ci) * 9 + t];
}

// ---------------------------------------------------------------------------
// 3x3 SAME conv + BN scale/bias + ReLU as per-tap implicit GEMM with WMMA.
// Grid: (W/32, H/8, B). Block: 256 threads (8 waves).
// Each WG: output tile 8 rows x 32 cols, all 64 out-channels.
// Wave w: co tile = (w&3)*16, pixel half = w>>2 (128 px = 8 N-tiles).
// B fragments are double-buffered so ds_load pairs stay one tile ahead of
// their consuming v_wmma (s_wait_dscnt > 0 instead of full serialization).
// Optionally also writes a bf16 copy in block layout [b][p][n][c] (for attn).
// ---------------------------------------------------------------------------
__global__ __launch_bounds__(256)
void k_conv3x3(const float* __restrict__ in, const bf16* __restrict__ wp,
               const float* __restrict__ sc, const float* __restrict__ bi,
               float* __restrict__ out, bf16* __restrict__ outT) {
  __shared__ bf16 sIn[10 * 34 * CC];  // input tile + halo, bf16, 43.5 KB
  const int b   = blockIdx.z;
  const int y0  = blockIdx.y * 8;
  const int x0  = blockIdx.x * 32;
  const int tid = threadIdx.x;

  // stage halo tile (zero padded), convert to bf16
  for (int f = tid; f < 10 * 34 * CC; f += 256) {
    int ci = f / 340, rem = f % 340, py = rem / 34, px = rem % 34;
    int gy = y0 + py - 1, gx = x0 + px - 1;
    float v = 0.f;
    if (gy >= 0 && gy < HH && gx >= 0 && gx < WW)
      v = in[((b * CC + ci) * HH + gy) * WW + gx];
    sIn[(py * 34 + px) * CC + ci] = (bf16)v;
  }
  __syncthreads();

  const int wave = tid >> 5, lane = tid & 31;
  const int half = lane >> 4, l16 = lane & 15;
  const int coB  = (wave & 3) * 16;
  const int ph   = wave >> 2;

  v8f acc[8];
#pragma unroll
  for (int i = 0; i < 8; ++i) acc[i] = vzero();

  for (int t = 0; t < 9; ++t) {
    const int dy = t / 3, dx = t % 3;
#pragma unroll
    for (int ks = 0; ks < 2; ++ks) {
      const int kb = ks * 32;
      // A fragment: W[co][ci], M=co, K=ci. K=16*half+e
      const bf16* ap = wp + t * (CC * CC) + (coB + l16) * CC + kb + 16 * half;
      v16bf A;
#pragma unroll
      for (int e = 0; e < 8; ++e) { A[e] = ap[e]; A[e + 8] = ap[8 + e]; }

      // double-buffered B fragments (K = 16*half + e, N = lane%16)
      const int pix0 = ph * 128 + l16;
      const int iy0 = (pix0 >> 5) + dy, ix0 = (pix0 & 31) + dx;
      const bf16* bp = sIn + (iy0 * 34 + ix0) * CC + kb + 16 * half;
      v16bf Bcur;
#pragma unroll
      for (int e = 0; e < 16; ++e) Bcur[e] = bp[e];
#pragma unroll
      for (int nt = 0; nt < 8; ++nt) {
        v16bf Bnext;
        if (nt < 7) {
          const int pix = ph * 128 + (nt + 1) * 16 + l16;
          const int iy = (pix >> 5) + dy, ix = (pix & 31) + dx;
          const bf16* bn = sIn + (iy * 34 + ix) * CC + kb + 16 * half;
#pragma unroll
          for (int e = 0; e < 16; ++e) Bnext[e] = bn[e];
        }
        acc[nt] = wmma_bf16(A, Bcur, acc[nt]);
        Bcur = Bnext;
      }
    }
  }

  // epilogue: BN + ReLU, store f32 NCHW (+ optional bf16 block layout)
  float scl[2], bia[2];
#pragma unroll
  for (int h = 0; h < 2; ++h) {
    // each lane only ever touches co = coB + r + 8*half for its own half
    scl[h] = 0.f; bia[h] = 0.f;  // overwritten below per r; kept simple
  }
#pragma unroll
  for (int nt = 0; nt < 8; ++nt) {
#pragma unroll
    for (int r = 0; r < 8; ++r) {
      const int co  = coB + r + 8 * half;           // M = r + 8*half
      const int pix = ph * 128 + nt * 16 + l16;     // N = lane%16
      const int gy = y0 + (pix >> 5), gx = x0 + (pix & 31);
      float v = acc[nt][r] * sc[co] + bi[co];
      v = v > 0.f ? v : 0.f;
      out[((b * CC + co) * HH + gy) * WW + gx] = v;
      if (outT) {
        const int p = (gy >> 5) * 8 + (gx >> 5);
        const int n = (gy & 31) * 32 + (gx & 31);
        outT[((size_t)(b * 64 + p) * 1024 + n) * CC + co] = (bf16)v;
      }
    }
  }
}

// ---------------------------------------------------------------------------
// Per (b,p): sy[n]=sum_c wy[c]*Y[c][n], sx[n]=sum_c wx[c]*X[c][n], + means.
// ---------------------------------------------------------------------------
__global__ __launch_bounds__(256)
void k_scores(const float* __restrict__ xc, const float* __restrict__ yc,
              const float* __restrict__ lnw, float* __restrict__ sy,
              float* __restrict__ sx, float* __restrict__ mY,
              float* __restrict__ mX) {
  __shared__ float redY[256], redX[256];
  const int bp = blockIdx.x;           // b*64 + p
  const int b = bp >> 6, p = bp & 63;
  const int y0 = (p >> 3) * 32, x0 = (p & 7) * 32;
  const int tid = threadIdx.x;
  float sumY = 0.f, sumX = 0.f;
  for (int j = 0; j < 4; ++j) {
    const int n = tid + j * 256;
    const int gy = y0 + (n >> 5), gx = x0 + (n & 31);
    float ay = 0.f, ax = 0.f;
    for (int c = 0; c < CC; ++c) {
      const int off = ((b * CC + c) * HH + gy) * WW + gx;
      ay += lnw[c] * yc[off];
      ax += lnw[CC + c] * xc[off];
    }
    sy[bp * 1024 + n] = ay;
    sx[bp * 1024 + n] = ax;
    sumY += ay; sumX += ax;
  }
  redY[tid] = sumY; redX[tid] = sumX;
  __syncthreads();
  for (int s = 128; s > 0; s >>= 1) {
    if (tid < s) { redY[tid] += redY[tid + s]; redX[tid] += redX[tid + s]; }
    __syncthreads();
  }
  if (tid == 0) {
    mY[bp] = redY[0] * (1.f / 1024.f);
    mX[bp] = redX[0] * (1.f / 1024.f);
  }
}

// ---------------------------------------------------------------------------
// Per (b,p1): score[p2] = mY[p2]+mX[p1]+2*bias + max_n(sy[p2][n]+sx[p1][n]),
// then top-5 over p2 (descending, first-index tie-break, like lax.top_k).
// ---------------------------------------------------------------------------
__global__ __launch_bounds__(256)
void k_topk(const float* __restrict__ sy, const float* __restrict__ sx,
            const float* __restrict__ mY, const float* __restrict__ mX,
            const float* __restrict__ lnb, int* __restrict__ idx) {
  __shared__ float red[256];
  __shared__ float scv[64];
  const int bp1 = blockIdx.x;
  const int b = bp1 >> 6;
  const int tid = threadIdx.x;
  const int p2 = tid >> 2, q = tid & 3;
  const float* syp = sy + (size_t)(b * 64 + p2) * 1024 + q * 256;
  const float* sxp = sx + (size_t)bp1 * 1024 + q * 256;
  float m = -3.4e38f;
  for (int n = 0; n < 256; ++n) {
    float v = syp[n] + sxp[n];
    m = v > m ? v : m;
  }
  red[tid] = m;
  __syncthreads();
  if (q == 0) {
    float m4 = fmaxf(fmaxf(red[tid], red[tid + 1]), fmaxf(red[tid + 2], red[tid + 3]));
    scv[p2] = mY[b * 64 + p2] + mX[bp1] + 2.f * lnb[0] + m4;
  }
  __syncthreads();
  if (tid == 0) {
    for (int k = 0; k < KSEL; ++k) {
      float best = -3.4e38f; int bsti = 0;
      for (int p = 0; p < 64; ++p)
        if (scv[p] > best) { best = scv[p]; bsti = p; }
      idx[bp1 * 8 + k] = bsti;
      scv[bsti] = -3.4e38f;
    }
  }
}

// ---------------------------------------------------------------------------
// Per (b,p1): G = sum_k Y_{pk} * X_{p1}^T  (64x64, K=1024)  [phase 1]
//             Out = G^T * X_{p1}           (64x1024, K=64)  [phase 2]
// Uses reassociated d = X^T (Ysum X^T): 16x fewer FLOPs than the reference
// and no 1024x1024 intermediate.
// ---------------------------------------------------------------------------
__global__ __launch_bounds__(256)
void k_attn(const float* __restrict__ xc, const float* __restrict__ yc,
            const bf16* __restrict__ xT, const int* __restrict__ idx,
            float* __restrict__ att) {
  __shared__ bf16 sGT[64 * 64];  // G^T bf16: sGT[c][c'] = G[c'][c]
  const int bp1 = blockIdx.x;
  const int b = bp1 >> 6, p1 = bp1 & 63;
  const int p1y = (p1 >> 3) * 32, p1x = (p1 & 7) * 32;
  const int tid = threadIdx.x;
  const int wave = tid >> 5, lane = tid & 31;
  const int half = lane >> 4, l16 = lane & 15;

  // ---- phase 1: 16 tiles of 16x16; wave w -> (mi = w>>1, ni = (w&1)*2 +{0,1})
  const int mi  = wave >> 1;
  const int ni0 = (wave & 1) * 2;
  v8f g0 = vzero(), g1 = vzero();
  for (int k = 0; k < KSEL; ++k) {
    const int pk = idx[bp1 * 8 + k];
    const int pky = (pk >> 3) * 32, pkx = (pk & 7) * 32;
    const float* arow = yc + ((size_t)(b * CC + (mi * 16 + l16)) * HH + pky) * WW + pkx;
    for (int kc = 0; kc < 32; ++kc) {     // n chunks of 32 (one block row each)
      const float* ap = arow + kc * WW;   // A: M=c' (Ysum row), K=n
      v16bf A;
#pragma unroll
      for (int e = 0; e < 8; ++e) {
        A[e]     = (bf16)ap[16 * half + e];
        A[e + 8] = (bf16)ap[16 * half + 8 + e];
      }
      // B: K=n, N=c  (reads X rows, contiguous along n)
      const float* bq0 = xc + ((size_t)(b * CC + (ni0 * 16 + l16)) * HH + p1y + kc) * WW
                            + p1x + 16 * half;
      const float* bq1 = bq0 + 16 * HH * WW;
      v16bf B0, B1;
#pragma unroll
      for (int e = 0; e < 16; ++e) { B0[e] = (bf16)bq0[e]; B1[e] = (bf16)bq1[e]; }
      g0 = wmma_bf16(A, B0, g0);
      g1 = wmma_bf16(A, B1, g1);
    }
  }
#pragma unroll
  for (int r = 0; r < 8; ++r) {
    const int cp = mi * 16 + r + 8 * half;              // c' (M)
    sGT[(ni0 * 16 + l16) * 64 + cp]       = (bf16)g0[r];
    sGT[((ni0 + 1) * 16 + l16) * 64 + cp] = (bf16)g1[r];
  }
  __syncthreads();

  // ---- phase 2: Out[c][n] = sum_c' G^T[c][c'] X[c'][n]
  const int mi2 = wave & 3;       // c tile
  const int nh  = wave >> 2;      // pixel half: 512 px = 32 N-tiles
  v16bf A0, A1;
  {
    const bf16* gp = sGT + (mi2 * 16 + l16) * 64;       // M=c, K=c' contiguous
#pragma unroll
    for (int e = 0; e < 8; ++e) {
      A0[e] = gp[16 * half + e];       A0[e + 8] = gp[16 * half + 8 + e];
      A1[e] = gp[32 + 16 * half + e];  A1[e + 8] = gp[40 + 16 * half + e];
    }
  }
  const bf16* xblk = xT + (size_t)(b * 64 + p1) * 1024 * 64;
  // double-buffer the global B stream so loads run ahead of the WMMAs
  v16bf B0c, B1c;
  {
    const bf16* bq = xblk + ((size_t)(nh * 512 + l16)) * 64 + 16 * half;
#pragma unroll
    for (int e = 0; e < 16; ++e) { B0c[e] = bq[e]; B1c[e] = bq[32 + e]; }
  }
  for (int nt = 0; nt < 32; ++nt) {
    v16bf B0n, B1n;
    if (nt < 31) {
      const bf16* bq = xblk + ((size_t)(nh * 512 + (nt + 1) * 16 + l16)) * 64 + 16 * half;
#pragma unroll
      for (int e = 0; e < 16; ++e) { B0n[e] = bq[e]; B1n[e] = bq[32 + e]; }
    }
    v8f acc = vzero();
    acc = wmma_bf16(A0, B0c, acc);
    acc = wmma_bf16(A1, B1c, acc);
#pragma unroll
    for (int r = 0; r < 8; ++r) {
      const int c   = mi2 * 16 + r + 8 * half;
      const int pix = nh * 512 + nt * 16 + l16;
      att[((size_t)(b * CC + c) * HH + p1y + (pix >> 5)) * WW + p1x + (pix & 31)] = acc[r];
    }
    B0c = B0n; B1c = B1n;
  }
}

// ---------------------------------------------------------------------------
extern "C" void kernel_launch(void* const* d_in, const int* in_sizes, int n_in,
                              void* d_out, int out_size, void* d_ws, size_t ws_size,
                              hipStream_t stream) {
  (void)in_sizes; (void)n_in; (void)out_size; (void)ws_size;
  const float* x   = (const float*)d_in[0];
  const float* y   = (const float*)d_in[1];
  const float* xw  = (const float*)d_in[2];
  const float* xs  = (const float*)d_in[3];
  const float* xbb = (const float*)d_in[4];
  const float* yw  = (const float*)d_in[5];
  const float* yss = (const float*)d_in[6];
  const float* ybb = (const float*)d_in[7];
  const float* lnw = (const float*)d_in[8];
  const float* lnb = (const float*)d_in[9];
  const float* ow  = (const float*)d_in[10];
  const float* oss = (const float*)d_in[11];
  const float* obb = (const float*)d_in[12];

  const size_t NPIX = (size_t)NB * CC * HH * WW;  // 8,388,608 elements
  float* xc  = (float*)d_ws;
  float* yc  = xc + NPIX;
  float* att = yc + NPIX;
  bf16*  xT  = (bf16*)(att + NPIX);               // [b][p][n][c] bf16
  float* sy  = (float*)(xT + NPIX);
  float* sx  = sy + (size_t)NB * 64 * 1024;
  float* mY  = sx + (size_t)NB * 64 * 1024;
  float* mX  = mY + NB * 64;
  int*   idx = (int*)(mX + NB * 64);
  bf16*  wpx = (bf16*)(idx + NB * 64 * 8);
  bf16*  wpy = wpx + 9 * CC * CC;
  bf16*  wpo = wpy + 9 * CC * CC;

  dim3 cgrid(WW / 32, HH / 8, NB);
  k_pack_w<<<144, 256, 0, stream>>>(xw, wpx);
  k_pack_w<<<144, 256, 0, stream>>>(yw, wpy);
  k_pack_w<<<144, 256, 0, stream>>>(ow, wpo);
  k_conv3x3<<<cgrid, 256, 0, stream>>>(x, wpx, xs, xbb, xc, xT);
  k_conv3x3<<<cgrid, 256, 0, stream>>>(y, wpy, yss, ybb, yc, (bf16*)nullptr);
  k_scores<<<NB * 64, 256, 0, stream>>>(xc, yc, lnw, sy, sx, mY, mX);
  k_topk<<<NB * 64, 256, 0, stream>>>(sy, sx, mY, mX, lnb, idx);
  k_attn<<<NB * 64, 256, 0, stream>>>(xc, yc, xT, idx, att);
  k_conv3x3<<<cgrid, 256, 0, stream>>>(att, wpo, oss, obb, (float*)d_out,
                                       (bf16*)nullptr);
}
